// NeighborList_46969762349284
// MI455X (gfx1250) — compile-verified, matrix operand order checked by reading, not run on hardware
//
#include <hip/hip_runtime.h>
#include <hip/hip_bf16.h>

// ---- problem constants (match reference) ----
#define NATOMS   8192
#define TPB      256                 // 8 wave32's per block
#define JITERS   8                   // j's per thread
#define CHUNK_J  (TPB * JITERS)      // 2048 j-slots per block
#define CHUNKS   4                   // ceil((NATOMS-1)/CHUNK_J)
#define NBLOCKS  (NATOMS * CHUNKS)   // 32768 partial counts (128 KB in d_ws)

typedef int v8i __attribute__((ext_vector_type(8)));

// Kernel 1: generate masked pairs (store-bandwidth bound) + per-block count.
// Indices (i,j) are recomputed analytically -> the 268 MB all_pairs input is
// never read; only the 268 MB output stream touches HBM.
__global__ __launch_bounds__(TPB) void nl_pairs_kernel(
    const float* __restrict__ coords,   // [NATOMS,3]
    const float* __restrict__ box,      // [3]
    const float* __restrict__ cutoff,   // [1]
    int*         __restrict__ out,      // [2P+1] int32 (pairs_out flat, then npairs)
    int*         __restrict__ ws_counts)// [NBLOCKS] per-block partial counts
{
    const int i      = blockIdx.y;
    const int chunk  = blockIdx.x;
    const int tid    = threadIdx.x;
    const int bid    = i * gridDim.x + chunk;
    const int jstart = i + 1 + chunk * CHUNK_J;

    __shared__ int wave_part[2 * (TPB / 32)];   // 2 half-wave sums per wave

    if (jstart >= NATOMS) {                     // block-uniform: nothing in this chunk
        if (tid == 0) ws_counts[bid] = 0;
        return;
    }

    // row-start pair offset: off(i) = i*(2N-1-i)/2   (max 33.5M, fits int32)
    const int pbase = (int)(((long long)i * (2 * NATOMS - 1 - i)) >> 1)
                    + (jstart - (i + 1));

    const float bx = box[0], by = box[1], bz = box[2];
    const float ibx = 1.0f / bx, iby = 1.0f / by, ibz = 1.0f / bz;
    const float cut  = cutoff[0];
    const float cut2 = cut * cut;
    const float cix = coords[3 * i + 0];
    const float ciy = coords[3 * i + 1];
    const float ciz = coords[3 * i + 2];

    // one pair = one 8B store; pack {x=i (low word), y=j (high word)} into u64
    // (little-endian layout matches int32[2] in memory), always 8B-aligned.
    unsigned long long* __restrict__ out2 = (unsigned long long*)out;

    int cnt = 0;
#pragma unroll
    for (int s = 0; s < JITERS; ++s) {
        const int j   = jstart + s * TPB + tid; // consecutive lanes -> consecutive pairs
        const bool inr = (j < NATOMS);
        const int jc  = inr ? j : i;            // safe clamped address, no divergent exit
        float dx = coords[3 * jc + 0] - cix;
        float dy = coords[3 * jc + 1] - ciy;
        float dz = coords[3 * jc + 2] - ciz;
        // minimum image: d -= box * rint(d/box)  (rint == round-half-even, matches jnp.round)
        dx = __builtin_fmaf(-bx, __builtin_rintf(dx * ibx), dx);
        dy = __builtin_fmaf(-by, __builtin_rintf(dy * iby), dy);
        dz = __builtin_fmaf(-bz, __builtin_rintf(dz * ibz), dz);
        const float dr2 = __builtin_fmaf(dx, dx, __builtin_fmaf(dy, dy, dz * dz));
        const bool hit = inr && (dr2 < cut2);
        cnt += hit ? 1 : 0;
        if (inr) {
            const unsigned int lo = hit ? (unsigned int)i : 0xFFFFFFFFu;
            const unsigned int hi = hit ? (unsigned int)j : 0xFFFFFFFFu;
            const unsigned long long pr =
                ((unsigned long long)hi << 32) | (unsigned long long)lo;
            // streaming output (never re-read): non-temporal, don't pollute 192MB L2
            __builtin_nontemporal_store(pr, &out2[pbase + s * TPB + tid]);
        }
    }

    // ---- wave-level count reduction on the matrix core (wave32, EXEC all 1s) ----
    // A (16x64 u8): one count byte per lane (lane m -> row m, lane m+16 -> row m,
    // different K). B = all-ones => D[m][n] = cnt(lane m) + cnt(lane m+16) for all n.
    v8i a = {};
    a[0] = cnt;                                 // cnt <= 8, byte 0 of lane's A row
    v8i b;
#pragma unroll
    for (int k = 0; k < 8; ++k) b[k] = 0x01010101;  // every u8 element = 1
    v8i c = {};
    c = __builtin_amdgcn_wmma_i32_16x16x64_iu8(false, a, false, b, c, false, false);
    const int s8 = c[0] + c[1] + c[2] + c[3] + c[4] + c[5] + c[6] + c[7];
    // lane 0 holds sum over rows 0..7 (= lanes 0..7 + 16..23), lane 16 rows 8..15
    const int lane = tid & 31;
    const int wave = tid >> 5;
    if (lane == 0)  wave_part[2 * wave + 0] = s8;
    if (lane == 16) wave_part[2 * wave + 1] = s8;
    __syncthreads();
    if (tid == 0) {
        int t = 0;
#pragma unroll
        for (int k = 0; k < 2 * (TPB / 32); ++k) t += wave_part[k];
        ws_counts[bid] = t;
    }
}

// Kernel 2: deterministic single-block reduction of the 32768 partial counts
// (no same-address global atomics anywhere -> no L2 serialization).
__global__ __launch_bounds__(1024) void nl_reduce_kernel(
    const int* __restrict__ ws_counts,
    int*       __restrict__ out)
{
    __shared__ int sm[1024];
    const int t = threadIdx.x;
    int acc = 0;
    for (int idx = t; idx < NBLOCKS; idx += 1024) acc += ws_counts[idx];
    sm[t] = acc;
    __syncthreads();
#pragma unroll
    for (int sft = 512; sft > 0; sft >>= 1) {
        if (t < sft) sm[t] += sm[t + sft];
        __syncthreads();
    }
    if (t == 0) {
        const long long np2 = (long long)NATOMS * (NATOMS - 1); // == 2*P
        out[np2] = sm[0];                                       // npairs slot
    }
}

extern "C" void kernel_launch(void* const* d_in, const int* in_sizes, int n_in,
                              void* d_out, int out_size, void* d_ws, size_t ws_size,
                              hipStream_t stream) {
    const float* coords = (const float*)d_in[0];   // [NATOMS,3] f32
    const float* box    = (const float*)d_in[1];   // [3] f32
    const float* cutoff = (const float*)d_in[2];   // [1] f32
    // d_in[3] (all_pairs, 268 MB) intentionally unused: indices are recomputed,
    // halving HBM traffic vs. reading the buffer.
    int* out       = (int*)d_out;
    int* ws_counts = (int*)d_ws;                   // needs NBLOCKS*4 = 128 KB scratch

    dim3 grid(CHUNKS, NATOMS, 1);
    hipLaunchKernelGGL(nl_pairs_kernel, grid, dim3(TPB), 0, stream,
                       coords, box, cutoff, out, ws_counts);
    hipLaunchKernelGGL(nl_reduce_kernel, dim3(1), dim3(1024), 0, stream,
                       ws_counts, out);
    (void)in_sizes; (void)n_in; (void)out_size; (void)ws_size;
}
